// GAT_11544872092126
// MI455X (gfx1250) — matrix operand order repaired and checked by reading, not compiled, not run on hardware
//
#include <hip/hip_runtime.h>
#include <hip/hip_bf16.h>

typedef __attribute__((ext_vector_type(16))) __bf16 v16bf;
typedef __attribute__((ext_vector_type(8)))  __bf16 v8bf;
typedef __attribute__((ext_vector_type(8)))  float  v8f;
typedef __attribute__((ext_vector_type(4)))  unsigned u32x4;
typedef __attribute__((ext_vector_type(8)))  int      i32x8;
typedef __attribute__((ext_vector_type(4)))  int      i32x4;

union Frag16 { v16bf v; v8bf h[2]; };

#define NEG_SLOPE 0.2f

// ---- Tensor Data Mover availability (toolchain-dependent arity) ----
#if defined(__has_builtin)
#if __has_builtin(__builtin_amdgcn_tensor_load_to_lds) && __has_builtin(__builtin_amdgcn_s_wait_tensorcnt)
#define HAVE_TDM 1
#endif
#endif
#ifndef HAVE_TDM
#define HAVE_TDM 0
#endif

__device__ __forceinline__ unsigned fkey(float f) {
  unsigned u = __float_as_uint(f);
  return (u & 0x80000000u) ? ~u : (u | 0x80000000u);
}
__device__ __forceinline__ float funkey(unsigned k) {
  unsigned u = (k & 0x80000000u) ? (k & 0x7fffffffu) : ~k;
  return __uint_as_float(u);
}

// ---------------- utility kernels ----------------
__global__ void k_fill_u32(unsigned* __restrict__ p, unsigned v, int n) {
  int t = blockIdx.x * blockDim.x + threadIdx.x;
  if (t < n) p[t] = v;
}
__global__ void k_fill_f32(float* __restrict__ p, float v, int n) {
  int t = blockIdx.x * blockDim.x + threadIdx.x;
  if (t < n) p[t] = v;
}
__global__ void k_cvt_bf16(const float* __restrict__ s, __bf16* __restrict__ d, int n) {
  int t = blockIdx.x * blockDim.x + threadIdx.x;
  if (t < n) d[t] = (__bf16)s[t];
}

// Pack row-major W[K][N] (f32) into per-lane WMMA B fragments (bf16).
// Packed element t = ((kt*NT + nt)*32 + lane)*16 + j  holds  W[kt*32 + j + 16*(lane>=16)][nt*16 + (lane&15)]
__global__ void k_pack_w(const float* __restrict__ W, __bf16* __restrict__ Bp, int K, int N) {
  int t = blockIdx.x * blockDim.x + threadIdx.x;
  if (t >= K * N) return;
  int NT   = N >> 4;
  int tile = t >> 9;
  int rem  = t & 511;
  int lane = rem >> 4;
  int j    = rem & 15;
  int kt = tile / NT;
  int nt = tile - kt * NT;
  int kk = kt * 32 + j + ((lane >= 16) ? 16 : 0);
  int nn = nt * 16 + (lane & 15);
  Bp[t] = (__bf16)W[(size_t)kk * N + nn];
}

// ---------------- WMMA GEMM: C[M,N] = act(A[M,K] @ B[K,N] + bias) ----------------
// Packed B is staged into LDS once per workgroup (TDM if available), then all 8
// waves consume it via ds_load_b128. One wave computes a full 16xN strip.
// OUTMODE: bit0 = write f32 C, bit1 = write bf16 C.
template<int KT, int NT, int OUTMODE>
__global__ __launch_bounds__(256) void k_gemm(
    const __bf16* __restrict__ A, const __bf16* __restrict__ Bp,
    const float* __restrict__ bias, float* __restrict__ Cf,
    __bf16* __restrict__ Cbf, int M, int doRelu)
{
  constexpr int K = KT * 32;
  constexpr int N = NT * 16;
  constexpr int BTOT = KT * NT * 512;          // bf16 elements of packed B
  __shared__ __bf16 Bs[BTOT];

#if HAVE_TDM
  if (threadIdx.x < 32) {
    // Build Tensor DMA Descriptor (D#) for a 1-row tile copy of BTOT*2 bytes.
    const unsigned lds_off = (unsigned)(uintptr_t)(&Bs[0]);      // LDS byte offset
    const unsigned long long ga = (unsigned long long)(uintptr_t)Bp;
    const int nunits = (BTOT * 2) / 4;                           // 4-byte units

    u32x4 g0 = {0u, 0u, 0u, 0u};
    g0[0] = 1u;                                      // count=1, is_restore=0
    g0[1] = lds_off;                                 // lds_addr
    g0[2] = (unsigned)(ga & 0xffffffffu);            // global_addr[31:0]
    g0[3] = (unsigned)((ga >> 32) & 0x01ffffffu)     // global_addr[56:32]
            | 0x80000000u;                           // type=2 ("image")

    i32x8 g1 = {0, 0, 0, 0, 0, 0, 0, 0};
    g1[0] = 2 << 16;                                 // workgroup_mask=0, data_size=4B
    g1[1] = (nunits & 0xffff) << 16;                 // tensor_dim0[15:0] @ bits63:48
    g1[2] = ((nunits >> 16) & 0xffff) | (1 << 16);   // tensor_dim0[31:16], tensor_dim1=1
    g1[3] = (nunits & 0xffff) << 16;                 // tile_dim0 @ bits127:112
    g1[4] = 1;                                       // tile_dim1=1, tile_dim2=0
    g1[5] = nunits;                                  // tensor_dim0_stride[31:0]

    i32x4 g2 = {0, 0, 0, 0};
    i32x4 g3 = {0, 0, 0, 0};
#if __clang_major__ >= 23
    i32x8 g4 = {0, 0, 0, 0, 0, 0, 0, 0};
    __builtin_amdgcn_tensor_load_to_lds(g0, g1, g2, g3, g4, 0);
#else
    __builtin_amdgcn_tensor_load_to_lds(g0, g1, g2, g3, 0);
#endif
    __builtin_amdgcn_s_wait_tensorcnt(0);
  }
#else
  {
    const uint4* src = (const uint4*)Bp;             // 8 bf16 per uint4
    uint4* dst = (uint4*)Bs;
    for (int i = threadIdx.x; i < BTOT / 8; i += blockDim.x) dst[i] = src[i];
  }
#endif
  __syncthreads();

  const int lane  = threadIdx.x & 31;
  const int mtile = blockIdx.x * (blockDim.x >> 5) + (threadIdx.x >> 5);
  if (mtile * 16 >= M) return;   // uniform per wave -> EXEC all-ones for WMMA

  const int row  = mtile * 16 + (lane & 15);
  const int koff = (lane >= 16) ? 16 : 0;

  v8f acc[NT];
#pragma unroll
  for (int nt = 0; nt < NT; ++nt)
#pragma unroll
    for (int r = 0; r < 8; ++r) acc[nt][r] = 0.f;

#pragma unroll
  for (int kt = 0; kt < KT; ++kt) {
    const int kbase = kt * 32;
    Frag16 a;
    a.h[0] = *(const v8bf*)(A + (size_t)row * K + kbase + koff);
    a.h[1] = *(const v8bf*)(A + (size_t)row * K + kbase + 8 + koff);
    const __bf16* bp = Bs + ((kt * NT) * 32 + lane) * 16;
#pragma unroll
    for (int nt = 0; nt < NT; ++nt) {
      Frag16 b;
      b.h[0] = *(const v8bf*)(bp + nt * 512);
      b.h[1] = *(const v8bf*)(bp + nt * 512 + 8);
      acc[nt] = __builtin_amdgcn_wmma_f32_16x16x32_bf16(
          false, a.v, false, b.v, (short)0, acc[nt], false, false);
    }
  }

  const int rbase = mtile * 16 + ((lane >= 16) ? 8 : 0);
  const int col   = lane & 15;
#pragma unroll
  for (int nt = 0; nt < NT; ++nt) {
    const int n = nt * 16 + col;
    const float bz = bias ? bias[n] : 0.f;
#pragma unroll
    for (int r = 0; r < 8; ++r) {
      float v = acc[nt][r] + bz;
      if (doRelu) v = fmaxf(v, 0.f);
      const size_t idx = (size_t)(rbase + r) * N + n;
      if (OUTMODE & 1) Cf[idx]  = v;
      if (OUTMODE & 2) Cbf[idx] = (__bf16)v;
    }
  }
}

// ---------------- per-node attention dots: s[n,h] = hp[n,h,:] . att_s[h,:] ----------------
__global__ void k_node_dots(const float* __restrict__ hp,
                            const float* __restrict__ as_, const float* __restrict__ ad_,
                            float* __restrict__ sarr, float* __restrict__ darr,
                            int N, int H)
{
  int wid  = (blockIdx.x * blockDim.x + threadIdx.x) >> 5;
  int lane = threadIdx.x & 31;
  if (wid >= N) return;
  int c0 = lane * 4;
  const float* row = hp + (size_t)wid * 128;
  float ps = 0.f, pd = 0.f;
#pragma unroll
  for (int j = 0; j < 4; ++j) {
    float v = row[c0 + j];
    ps += v * as_[c0 + j];
    pd += v * ad_[c0 + j];
  }
  int gsz = 32 / H;                       // 16 (H=2) or 32 (H=1)
  for (int m = 1; m < gsz; m <<= 1) {
    ps += __shfl_xor(ps, m, 32);
    pd += __shfl_xor(pd, m, 32);
  }
  if ((lane & (gsz - 1)) == 0) {
    int h = lane / gsz;
    sarr[(size_t)wid * H + h] = ps;
    darr[(size_t)wid * H + h] = pd;
  }
}

// ---------------- edge pass 1: logits + segment max ----------------
__global__ void k_edge_logits(const int* __restrict__ snd, const int* __restrict__ rcv,
                              const float* __restrict__ sarr, const float* __restrict__ darr,
                              float* __restrict__ logits, unsigned* __restrict__ maxk,
                              int E, int H)
{
  int t = blockIdx.x * blockDim.x + threadIdx.x;
  if (t >= E * H) return;
  int e = t / H, h = t - e * H;
  int s = snd[e], r = rcv[e];
  float l = sarr[(size_t)s * H + h] + darr[(size_t)r * H + h];
  l = (l > 0.f) ? l : NEG_SLOPE * l;
  logits[t] = l;
  atomicMax(maxk + (size_t)r * H + h, fkey(l));
}

// ---------------- edge pass 2: exp + segment sum ----------------
__global__ void k_edge_expsum(const int* __restrict__ rcv,
                              const float* __restrict__ logits, const unsigned* __restrict__ maxk,
                              float* __restrict__ expv, float* __restrict__ sumb,
                              int E, int H)
{
  int t = blockIdx.x * blockDim.x + threadIdx.x;
  if (t >= E * H) return;
  int e = t / H, h = t - e * H;
  int r = rcv[e];
  float m  = funkey(maxk[(size_t)r * H + h]);
  float ev = __expf(logits[t] - m);
  expv[t] = ev;
  atomicAdd(sumb + (size_t)r * H + h, ev);
}

// ---------------- edge pass 3: weighted message scatter (one wave per edge) ----------------
__global__ void k_edge_scatter(const int* __restrict__ snd, const int* __restrict__ rcv,
                               const float* __restrict__ hp, const float* __restrict__ expv,
                               const float* __restrict__ sumb, float* __restrict__ gatout,
                               int E, int H, int dshift)
{
  int wid  = (blockIdx.x * blockDim.x + threadIdx.x) >> 5;
  int lane = threadIdx.x & 31;
  if (wid >= E) return;
  int s = snd[wid], r = rcv[wid];
  int c0 = lane * 4;
  int h  = c0 >> dshift;                          // 6 -> D=64 (H=2), 7 -> D=128 (H=1)
  float alpha = expv[(size_t)wid * H + h] / sumb[(size_t)r * H + h];
  const float* src = hp + (size_t)s * 128 + c0;
  float* dst = gatout + (size_t)r * 128 + c0;
#pragma unroll
  for (int j = 0; j < 4; ++j) atomicAdd(dst + j, src[j] * alpha);
}

// ---------------- combine: h = relu(gat) + skip; write f32 + bf16 ----------------
__global__ void k_combine(const float* __restrict__ gat, const float* __restrict__ skip,
                          float* __restrict__ hf, __bf16* __restrict__ hbf, int n)
{
  int t = blockIdx.x * blockDim.x + threadIdx.x;
  if (t >= n) return;
  float v = fmaxf(gat[t], 0.f) + skip[t];
  hf[t]  = v;
  hbf[t] = (__bf16)v;
}

// ---------------- segment-mean pooling (contiguous graphs of `per` nodes) ----------------
__global__ void k_pool(const float* __restrict__ hf, __bf16* __restrict__ pooledbf,
                       int G, int per)
{
  int g = blockIdx.x;       // launched with Mpad blocks; pads write zero
  int c = threadIdx.x;      // 128 threads
  float sum = 0.f;
  if (g < G) {
    const float* base = hf + (size_t)g * per * 128 + c;
    for (int i = 0; i < per; ++i) sum += base[(size_t)i * 128];
    sum /= (float)per;
  }
  pooledbf[(size_t)g * 128 + c] = (__bf16)sum;
}

// ---------------- final projection to scalar per graph ----------------
__global__ void k_final(const float* __restrict__ z, const float* __restrict__ w3,
                        const float* __restrict__ b3, float* __restrict__ out, int G)
{
  int g = blockIdx.x * blockDim.x + threadIdx.x;
  if (g >= G) return;
  float acc = b3[0];
  for (int c = 0; c < 128; ++c) acc += z[(size_t)g * 128 + c] * w3[c];
  out[g] = acc;
}

// =========================================================================
extern "C" void kernel_launch(void* const* d_in, const int* in_sizes, int n_in,
                              void* d_out, int out_size, void* d_ws, size_t ws_size,
                              hipStream_t stream) {
  const int N = 50000, G = 100, PER = 500, MPAD = 112;

  // ---- resolve inputs ----
  const float* x = (const float*)d_in[0];
  const float *w_in, *b_in, *wg[3], *as_[3], *ad_[3], *wsk[3], *bs_[3];
  const float *w1, *b1, *w2, *b2, *w3, *b3;
  const int *snd, *rcv;
  int E;

  if (n_in >= 27) {                 // params flattened to separate leaves (insertion order)
    int k = 1;
    w_in = (const float*)d_in[k++]; b_in = (const float*)d_in[k++];
    for (int i = 0; i < 3; ++i) {
      wg[i]  = (const float*)d_in[k++];
      as_[i] = (const float*)d_in[k++];
      ad_[i] = (const float*)d_in[k++];
      wsk[i] = (const float*)d_in[k++];
      bs_[i] = (const float*)d_in[k++];
    }
    w1 = (const float*)d_in[k++]; b1 = (const float*)d_in[k++];
    w2 = (const float*)d_in[k++]; b2 = (const float*)d_in[k++];
    w3 = (const float*)d_in[k++]; b3 = (const float*)d_in[k++];
    snd = (const int*)d_in[24]; rcv = (const int*)d_in[25];
    E = in_sizes[24];
  } else {                          // params as one concatenated buffer
    const float* p = (const float*)d_in[1];
    w_in = p; p += 64 * 64;  b_in = p; p += 64;
    for (int i = 0; i < 3; ++i) {
      int ind = (i == 0) ? 64 : 128;
      wg[i]  = p; p += (size_t)ind * 128;
      as_[i] = p; p += 128;
      ad_[i] = p; p += 128;
      wsk[i] = p; p += (size_t)ind * 128;
      bs_[i] = p; p += 128;
    }
    w1 = p; p += 128 * 128; b1 = p; p += 128;
    w2 = p; p += 128 * 128; b2 = p; p += 128;
    w3 = p; p += 128;       b3 = p; p += 1;
    snd = (const int*)d_in[2]; rcv = (const int*)d_in[3];
    E = in_sizes[2];
  }

  // ---- workspace carve-out (256B aligned) ----
  char* wp = (char*)d_ws;
  auto carve = [&](size_t bytes) -> void* {
    void* q = (void*)wp;
    wp += (bytes + 255) & ~(size_t)255;
    return q;
  };
  __bf16*  xbf    = (__bf16*)carve((size_t)N * 64 * 2);
  __bf16*  hbf    = (__bf16*)carve((size_t)N * 128 * 2);
  float*   hf     = (float*) carve((size_t)N * 128 * 4);
  float*   hpf    = (float*) carve((size_t)N * 128 * 4);
  float*   skipf  = (float*) carve((size_t)N * 128 * 4);
  float*   gatout = (float*) carve((size_t)N * 128 * 4);
  float*   sarr   = (float*) carve((size_t)N * 2 * 4);
  float*   darr   = (float*) carve((size_t)N * 2 * 4);
  unsigned* maxk  = (unsigned*)carve((size_t)N * 2 * 4);
  float*   sumb   = (float*) carve((size_t)N * 2 * 4);
  float*   logits = (float*) carve((size_t)E * 2 * 4);
  float*   expv   = (float*) carve((size_t)E * 2 * 4);
  __bf16*  pooledbf = (__bf16*)carve((size_t)MPAD * 128 * 2);
  float*   z1f    = (float*) carve((size_t)MPAD * 128 * 4);
  __bf16*  z1bf   = (__bf16*)carve((size_t)MPAD * 128 * 2);
  float*   z2f    = (float*) carve((size_t)MPAD * 128 * 4);
  __bf16*  w_in_p = (__bf16*)carve(64 * 64 * 2);
  __bf16*  wg_p0  = (__bf16*)carve(64 * 128 * 2);
  __bf16*  ws_p0  = (__bf16*)carve(64 * 128 * 2);
  __bf16*  wg_p1  = (__bf16*)carve(128 * 128 * 2);
  __bf16*  ws_p1  = (__bf16*)carve(128 * 128 * 2);
  __bf16*  wg_p2  = (__bf16*)carve(128 * 128 * 2);
  __bf16*  ws_p2  = (__bf16*)carve(128 * 128 * 2);
  __bf16*  w1_p   = (__bf16*)carve(128 * 128 * 2);
  __bf16*  w2_p   = (__bf16*)carve(128 * 128 * 2);
  __bf16* wg_p[3] = { wg_p0, wg_p1, wg_p2 };
  __bf16* ws_p[3] = { ws_p0, ws_p1, ws_p2 };

  auto cdiv = [](int a, int b) { return (a + b - 1) / b; };

  // ---- weight packing + input conversion ----
  k_pack_w<<<cdiv(64 * 64, 256),   256, 0, stream>>>(w_in, w_in_p, 64, 64);
  k_pack_w<<<cdiv(64 * 128, 256),  256, 0, stream>>>(wg[0], wg_p[0], 64, 128);
  k_pack_w<<<cdiv(64 * 128, 256),  256, 0, stream>>>(wsk[0], ws_p[0], 64, 128);
  for (int i = 1; i < 3; ++i) {
    k_pack_w<<<cdiv(128 * 128, 256), 256, 0, stream>>>(wg[i], wg_p[i], 128, 128);
    k_pack_w<<<cdiv(128 * 128, 256), 256, 0, stream>>>(wsk[i], ws_p[i], 128, 128);
  }
  k_pack_w<<<cdiv(128 * 128, 256), 256, 0, stream>>>(w1, w1_p, 128, 128);
  k_pack_w<<<cdiv(128 * 128, 256), 256, 0, stream>>>(w2, w2_p, 128, 128);
  k_cvt_bf16<<<cdiv(N * 64, 256), 256, 0, stream>>>(x, xbf, N * 64);

  const int gemmBlocksN = cdiv(N / 16, 8);   // 3125 M-tiles -> 391 blocks of 8 waves

  // ---- input layer: h0 = relu(x @ w_in + b_in) -> bf16 [N,64] ----
  k_gemm<2, 4, 2><<<gemmBlocksN, 256, 0, stream>>>(xbf, w_in_p, b_in, (float*)nullptr, hbf, N, 1);

  // ---- 3 GAT layers ----
  for (int i = 0; i < 3; ++i) {
    const int H = (i == 0) ? 2 : 1;
    const int dshift = (i == 0) ? 6 : 7;

    if (i == 0) {
      k_gemm<2, 8, 1><<<gemmBlocksN, 256, 0, stream>>>(hbf, wg_p[0], (const float*)nullptr, hpf, (__bf16*)nullptr, N, 0);
      k_gemm<2, 8, 1><<<gemmBlocksN, 256, 0, stream>>>(hbf, ws_p[0], bs_[0], skipf, (__bf16*)nullptr, N, 0);
    } else {
      k_gemm<4, 8, 1><<<gemmBlocksN, 256, 0, stream>>>(hbf, wg_p[i], (const float*)nullptr, hpf, (__bf16*)nullptr, N, 0);
      k_gemm<4, 8, 1><<<gemmBlocksN, 256, 0, stream>>>(hbf, ws_p[i], bs_[i], skipf, (__bf16*)nullptr, N, 0);
    }

    k_node_dots<<<cdiv(N * 32, 256), 256, 0, stream>>>(hpf, as_[i], ad_[i], sarr, darr, N, H);

    k_fill_u32<<<cdiv(N * H, 256), 256, 0, stream>>>(maxk, 0x007FFFFFu /* fkey(-inf) */, N * H);
    k_fill_f32<<<cdiv(N * H, 256), 256, 0, stream>>>(sumb, 0.f, N * H);
    k_fill_f32<<<cdiv(N * 128, 256), 256, 0, stream>>>(gatout, 0.f, N * 128);

    k_edge_logits<<<cdiv(E * H, 256), 256, 0, stream>>>(snd, rcv, sarr, darr, logits, maxk, E, H);
    k_edge_expsum<<<cdiv(E * H, 256), 256, 0, stream>>>(rcv, logits, maxk, expv, sumb, E, H);
    k_edge_scatter<<<cdiv(E * 32, 256), 256, 0, stream>>>(snd, rcv, hpf, expv, sumb, gatout, E, H, dshift);

    k_combine<<<cdiv(N * 128, 256), 256, 0, stream>>>(gatout, skipf, hf, hbf, N * 128);
  }

  // ---- pooling + MLP head ----
  k_pool<<<MPAD, 128, 0, stream>>>(hf, pooledbf, G, PER);
  k_gemm<4, 8, 3><<<cdiv(MPAD / 16, 8), 256, 0, stream>>>(pooledbf, w1_p, b1, z1f, z1bf, MPAD, 1);
  k_gemm<4, 8, 1><<<cdiv(MPAD / 16, 8), 256, 0, stream>>>(z1bf, w2_p, b2, z2f, (__bf16*)nullptr, MPAD, 1);
  k_final<<<cdiv(G, 128), 128, 0, stream>>>(z2f, w3, b3, (float*)d_out, G);

  (void)in_sizes; (void)n_in; (void)out_size; (void)ws_size;
}
